// CRF_80436147519573
// MI455X (gfx1250) — compile-verified, matrix operand order checked by reading, not run on hardware
//
#include <hip/hip_runtime.h>
#include <hip/hip_bf16.h>
#include <math.h>

typedef _Float16 v16h __attribute__((ext_vector_type(16)));
typedef _Float16 v8h  __attribute__((ext_vector_type(8)));
typedef float    v8f  __attribute__((ext_vector_type(8)));

constexpr int kS = 512;     // sequence length
constexpr int kB = 256;     // batch
constexpr int kL = 128;     // labels
constexpr int kPAD = 0;
constexpr int kSTART = 1;
constexpr int kAStride = 136;   // halves per A row in LDS: 128 + 8 pad -> bank offset 4/row, 16B aligned
constexpr float kLN2 = 0.69314718055994530942f;

// ---------------------------------------------------------------------------
// Forward (partition function) kernel: one block = 16 batch rows, 8 waves.
// Scaled linear-domain recurrence: per row keep offset mr and u = exp(fs - mr).
//   normalize: umax = max_n u ; p = u * rcp(umax) ; mr += ln(umax)
//   GEMM:      C(16x16) = P(16x128) x E(128x16), E = exp(T) resident in VGPRs,
//              4 chained v_wmma_f32_16x16x32_f16
//   update:    u' = C * exp(emit[t])  (exp pipelined one step ahead)
// Only 1 log + 1 rcp per row per step on the critical path. 2 barriers/step.
// ---------------------------------------------------------------------------
__launch_bounds__(256)
__global__ void crf_forward_kernel(const float* __restrict__ emit,
                                   const unsigned char* __restrict__ masks,
                                   const float* __restrict__ T,
                                   float* __restrict__ ws_encode)
{
    __shared__ _Float16 As[16 * kAStride];   // P = exp(fs - mr), f16
    __shared__ float    wmax[8][16];         // per-wave row maxima
    __shared__ float    wsum[8][16];         // per-wave row partial sums (epilogue)
    __shared__ float    mrow[16];            // combined row max (epilogue)
    __shared__ int      lenS[16];            // sequence lengths for this row tile

    const int tid  = threadIdx.x;
    const int wave = tid >> 5;
    const int lane = tid & 31;
    const int lo   = lane & 15;        // N index within wave tile / M for A-frag
    const int hi   = lane >> 4;        // lane-half selector
    const int n0   = wave * 16;        // this wave's column block
    const int n    = n0 + lo;          // global label/column index for this lane
    const int b0   = blockIdx.x * 16;  // first batch row of this block

    // ---- lengths from mask bytes (masks are prefix-of-ones; count suffices) ----
    if (tid < 16) lenS[tid] = 0;
    __syncthreads();
    {
        const int row = tid >> 4;      // 0..15
        const int seg = tid & 15;      // 16 threads x 32 bytes = 512
        const unsigned char* mp = masks + (size_t)(b0 + row) * kS + seg * 32;
        int c = 0;
        #pragma unroll
        for (int i = 0; i < 32; ++i) c += (mp[i] != 0) ? 1 : 0;
        atomicAdd(&lenS[row], c);
    }
    __syncthreads();
    int lenR[8];
    #pragma unroll
    for (int r = 0; r < 8; ++r) lenR[r] = lenS[r + 8 * hi];

    // ---- E = exp(T) as resident WMMA B fragments (32x16 f16 per K-block) ----
    // B layout (16-bit, 32x16): lane: N = lane&15; elem i: K = 8*(lane>>4) + i + (i>=8 ? 8 : 0)
    v16h bf[4];
    #pragma unroll
    for (int kf = 0; kf < 4; ++kf) {
        #pragma unroll
        for (int i = 0; i < 16; ++i) {
            const int K = kf * 32 + hi * 8 + i + ((i >= 8) ? 8 : 0);
            bf[kf][i] = (_Float16)__expf(T[K * kL + n]);
        }
    }

    // ---- t = 0:  u = exp(fs0) = exp(emit[0] + T[START]),  mr = 0 ----
    // C/D layout: lane holds (m = r + 8*hi, n) for accumulator register r.
    float u[8], mrR[8];
    const float tstart = T[kSTART * kL + n];
    #pragma unroll
    for (int r = 0; r < 8; ++r) {
        u[r]   = __expf(emit[(unsigned)(b0 + r + 8 * hi) * kL + n] + tstart);
        mrR[r] = 0.f;
    }

    // ---- pipeline prologue: exp(emit) for t = 1 ----
    float ee[8];
    {
        const unsigned ebase = (unsigned)(1 * kB + b0) * kL + (unsigned)n;
        #pragma unroll
        for (int r = 0; r < 8; ++r)
            ee[r] = __expf(emit[ebase + (unsigned)(r + 8 * hi) * kL]);
    }

    // ---- forward recurrence (2 barriers per step) ----
    for (int t = 1; t < kS; ++t) {
        // (0) software-pipelined emit loads for step t+1
        float enxt[8];
        if (t + 1 < kS) {
            const unsigned ebase = (unsigned)((t + 1) * kB + b0) * kL + (unsigned)n;
            #pragma unroll
            for (int r = 0; r < 8; ++r) enxt[r] = emit[ebase + (unsigned)(r + 8 * hi) * kL];
            __builtin_prefetch(emit + (size_t)ebase + (size_t)kB * kL, 0, 1);  // t+2
        }

        // (1) per-wave row max of u over this wave's 16 columns (butterfly per half)
        #pragma unroll
        for (int r = 0; r < 8; ++r) {
            float v = u[r];
            v = fmaxf(v, __shfl_xor(v, 1, 32));
            v = fmaxf(v, __shfl_xor(v, 2, 32));
            v = fmaxf(v, __shfl_xor(v, 4, 32));
            v = fmaxf(v, __shfl_xor(v, 8, 32));
            if (lo == 0) wmax[wave][r + 8 * hi] = v;
        }
        __syncthreads();

        // (2) cross-wave combine, redundantly per lane (row = lo):
        //     one log + one rcp per row, broadcast via shuffle.
        float um = wmax[0][lo];
        #pragma unroll
        for (int w = 1; w < 8; ++w) um = fmaxf(um, wmax[w][lo]);
        const float lgv = kLN2 * __log2f(um);       // ln(umax)
        const float rcv = __builtin_amdgcn_rcpf(um); // 1/umax (feeds f16, approx ok)
        float lgs[8], rcs[8];
        #pragma unroll
        for (int r = 0; r < 8; ++r) {
            lgs[r] = __shfl(lgv, r + 8 * hi, 32);
            rcs[r] = __shfl(rcv, r + 8 * hi, 32);
        }

        // (3) normalize (pure bookkeeping, freeze-safe): p = u/umax in (0,1],
        //     store to As (f16, padded stride), advance row offset.
        float pr[8];
        #pragma unroll
        for (int r = 0; r < 8; ++r) {
            const float p = u[r] * rcs[r];
            pr[r] = p;
            As[(r + 8 * hi) * kAStride + n] = (_Float16)p;
            mrR[r] += lgs[r];
        }
        __syncthreads();

        // (4) A fragments + chained WMMAs
        // A layout (16-bit, 16x32): lane: M = lane&15; elem i: K = kf*32 + 8*(lane>>4) + i + (i>=8 ? 8 : 0)
        v16h a[4];
        #pragma unroll
        for (int kf = 0; kf < 4; ++kf) {
            const int basek = kf * 32 + hi * 8;
            const v8h a_lo = *(const v8h*)&As[lo * kAStride + basek];
            const v8h a_hi = *(const v8h*)&As[lo * kAStride + basek + 16];
            a[kf] = __builtin_shufflevector(a_lo, a_hi,
                                            0,1,2,3,4,5,6,7,8,9,10,11,12,13,14,15);
        }
        v8f c = {};
        #pragma unroll
        for (int kf = 0; kf < 4; ++kf) {
            c = __builtin_amdgcn_wmma_f32_16x16x32_f16(
                    false, a[kf], false, bf[kf], (short)0, c, false, false);
        }

        // (5) branchless linear-domain update: u' = C * exp(emit) if active,
        //     else keep (normalized) old value.
        #pragma unroll
        for (int r = 0; r < 8; ++r) {
            u[r] = (t < lenR[r]) ? c[r] * ee[r] : pr[r];
        }

        // (6) rotate pipeline: exponentiate next step's emit off the critical path
        if (t + 1 < kS) {
            #pragma unroll
            for (int r = 0; r < 8; ++r) ee[r] = __expf(enxt[r]);
        }
    }

    // ---- epilogue: back to log domain, encode_partial = sum_m logsumexp_n(fs + T[n][PAD]) ----
    const float tpad = T[n * kL + kPAD];
    float val[8];
    #pragma unroll
    for (int r = 0; r < 8; ++r) val[r] = mrR[r] + kLN2 * __log2f(u[r]) + tpad;

    #pragma unroll
    for (int r = 0; r < 8; ++r) {
        float v = val[r];
        v = fmaxf(v, __shfl_xor(v, 1, 32));
        v = fmaxf(v, __shfl_xor(v, 2, 32));
        v = fmaxf(v, __shfl_xor(v, 4, 32));
        v = fmaxf(v, __shfl_xor(v, 8, 32));
        if (lo == 0) wmax[wave][r + 8 * hi] = v;
    }
    __syncthreads();
    if (tid < 16) {
        float v = wmax[0][tid];
        #pragma unroll
        for (int w = 1; w < 8; ++w) v = fmaxf(v, wmax[w][tid]);
        mrow[tid] = v;
    }
    __syncthreads();
    #pragma unroll
    for (int r = 0; r < 8; ++r) {
        const int m = r + 8 * hi;
        float v = __expf(val[r] - mrow[m]);
        v += __shfl_xor(v, 1, 32);
        v += __shfl_xor(v, 2, 32);
        v += __shfl_xor(v, 4, 32);
        v += __shfl_xor(v, 8, 32);
        if (lo == 0) wsum[wave][m] = v;
    }
    __syncthreads();
    if (tid == 0) {
        float acc = 0.f;
        for (int m = 0; m < 16; ++m) {
            float s = 0.f;
            #pragma unroll
            for (int w = 0; w < 8; ++w) s += wsum[w][m];
            acc += mrow[m] + kLN2 * __log2f(s);
        }
        ws_encode[blockIdx.x] = acc;
    }
}

// ---------------------------------------------------------------------------
// Gold path score: one thread per batch row (tiny gather/sum).
// ---------------------------------------------------------------------------
__launch_bounds__(256)
__global__ void crf_gold_kernel(const float* __restrict__ emit,
                                const int* __restrict__ labels,
                                const unsigned char* __restrict__ masks,
                                const float* __restrict__ T,
                                float* __restrict__ ws_gold)
{
    const int b = blockIdx.x * blockDim.x + threadIdx.x;
    if (b >= kB) return;
    const unsigned char* mp = masks + (size_t)b * kS;
    int len = 0;
    for (int t = 0; t < kS; ++t) len += (mp[t] != 0) ? 1 : 0;

    const int* lab = labels + (size_t)b * kS;
    float g = 0.f;
    int prev = kSTART;
    for (int t = 0; t < len; ++t) {
        const int nx = lab[t];
        g += emit[((size_t)t * kB + b) * kL + nx] + T[prev * kL + nx];
        prev = nx;
    }
    g += T[prev * kL + kPAD];   // prev == labels[b][len-1]  (len >= 1)
    ws_gold[b] = g;
}

// ---------------------------------------------------------------------------
// Final deterministic combine: out = encode - gold  (scalar).
// ---------------------------------------------------------------------------
__global__ void crf_final_kernel(const float* __restrict__ ws, float* __restrict__ out)
{
    if (threadIdx.x == 0 && blockIdx.x == 0) {
        float enc = 0.f;
        for (int i = 0; i < kB / 16; ++i) enc += ws[i];
        float gold = 0.f;
        for (int i = 0; i < kB; ++i) gold += ws[kB / 16 + i];
        out[0] = enc - gold;
    }
}

extern "C" void kernel_launch(void* const* d_in, const int* in_sizes, int n_in,
                              void* d_out, int out_size, void* d_ws, size_t ws_size,
                              hipStream_t stream)
{
    const float*         emit   = (const float*)d_in[0];          // (S,B,L) f32
    const int*           labels = (const int*)d_in[1];            // (B,S) i32
    const unsigned char* masks  = (const unsigned char*)d_in[2];  // (B,S) bool (1 byte)
    const float*         T      = (const float*)d_in[3];          // (L,L) f32
    float* ws = (float*)d_ws;   // [0..15]: encode partials, [16..271]: gold per batch

    crf_forward_kernel<<<kB / 16, 256, 0, stream>>>(emit, masks, T, ws);
    crf_gold_kernel<<<1, 256, 0, stream>>>(emit, labels, masks, T, ws + kB / 16);
    crf_final_kernel<<<1, 1, 0, stream>>>(ws, (float*)d_out);
}
